// MPNN_13348758356092
// MI455X (gfx1250) — matrix operand order repaired and checked by reading, not compiled, not run on hardware
//
#include <hip/hip_runtime.h>
#include <hip/hip_bf16.h>
#include <math.h>

typedef __attribute__((ext_vector_type(16))) _Float16 v16h;
typedef __attribute__((ext_vector_type(8)))  _Float16 v8h;
typedef __attribute__((ext_vector_type(8)))  float    v8f;

#define HID 32
#define TPW 4   // 16-edge tiles per wave (B matrix stays register-resident)

// ---- monotonic order-preserving float<->u32 keys (branchless atomic max) ----
__device__ __forceinline__ unsigned f2key(float f) {
  unsigned b = __float_as_uint(f);
  return (b & 0x80000000u) ? ~b : (b | 0x80000000u);
}
__device__ __forceinline__ float key2f(unsigned k) {
  return (k & 0x80000000u) ? __uint_as_float(k ^ 0x80000000u) : __uint_as_float(~k);
}
#define KEY_NEG_INF 0x007FFFFFu  // f2key(-inf)

// ---------------------------------------------------------------------------
// One-time setup: hidden=0, preds row0 = states[0], stop[0]=0
// ---------------------------------------------------------------------------
__global__ void k_setup(const float* __restrict__ states, float* __restrict__ hidden,
                        float* __restrict__ preds, float* __restrict__ stops, int N) {
  int i = blockIdx.x * blockDim.x + threadIdx.x;
  if (i < N * HID) hidden[i] = 0.f;
  if (i < N) preds[i] = states[i];
  if (i == 0) stops[0] = 0.f;
}

// ---------------------------------------------------------------------------
// One-time: edge_feat[e] = edges_mat[src[e], dst[e]]  (400MB gather, f32 out)
// ---------------------------------------------------------------------------
__global__ void k_edgefeat(const float* __restrict__ edges_mat, const int* __restrict__ src,
                           const int* __restrict__ dst, float* __restrict__ ef,
                           long E, int N) {
  long e = (long)blockIdx.x * blockDim.x + threadIdx.x;
  if (e >= E) return;
  ef[e] = edges_mat[(size_t)src[e] * (size_t)N + (size_t)dst[e]];
}

// ---------------------------------------------------------------------------
// One-time: pack M_w rows 0..63 (f32) into f16 WMMA B operands (contiguous k order, same as A), 2 K-chunks x
// 2 N-tiles. Slot = (chunk*2 + tile)*32 + lane; 16 halves per slot:
//   element i of lane -> B[ chunk*32 + (lane>>4)*16 + i ][ tile*16 + (lane&15) ]
// (row 64, the edge-feature weight, is applied in f32 in the edge epilogue)
// ---------------------------------------------------------------------------
__global__ void k_packB(const float* __restrict__ M_w, _Float16* __restrict__ Bh) {
  int slot = threadIdx.x;
  if (slot >= 2 * 2 * 32) return;
  int chunk = slot / 64, rem = slot % 64;
  int tile = rem / 32, lane = rem % 32;
  int hh = lane >> 4, ncol = tile * 16 + (lane & 15);
  for (int i = 0; i < 16; ++i) {
    int K = chunk * 32 + hh * 16 + i;
    Bh[slot * 16 + i] = (_Float16)M_w[K * HID + ncol];
  }
}

// ---------------------------------------------------------------------------
// Per-step encode: z = [state, hidden, pri] @ enc_w + enc_b  (f32 VALU)
// also: zh = f16(z); u_keys = key(-inf); reset hsum/locmax for this step
// ---------------------------------------------------------------------------
__global__ void k_encode(const float* __restrict__ state, const float* __restrict__ hidden,
                         const float* __restrict__ pri,
                         const float* __restrict__ enc_w, const float* __restrict__ enc_b,
                         float* __restrict__ z, _Float16* __restrict__ zh,
                         unsigned* __restrict__ ukeys, float* __restrict__ hsum,
                         unsigned int* __restrict__ locmax, int N) {
  __shared__ float Wl[34 * HID];
  __shared__ float Bl[HID];
  for (int i = threadIdx.x; i < 34 * HID; i += blockDim.x) Wl[i] = enc_w[i];
  if (threadIdx.x < HID) Bl[threadIdx.x] = enc_b[threadIdx.x];
  __syncthreads();
  int n = blockIdx.x * blockDim.x + threadIdx.x;
  if (n >= N) return;
  float inp[34];
  inp[0] = state[n];
#pragma unroll
  for (int k = 0; k < HID; ++k) inp[1 + k] = hidden[n * HID + k];
  inp[33] = pri[n];
  float acc[HID];
#pragma unroll
  for (int c = 0; c < HID; ++c) acc[c] = Bl[c];
  for (int k = 0; k < 34; ++k) {
    float v = inp[k];
#pragma unroll
    for (int c = 0; c < HID; ++c) acc[c] = fmaf(v, Wl[k * HID + c], acc[c]);
  }
#pragma unroll
  for (int c = 0; c < HID; ++c) {
    z[n * HID + c] = acc[c];
    zh[n * HID + c] = (_Float16)acc[c];
    ukeys[n * HID + c] = KEY_NEG_INF;
  }
  if (n < HID) hsum[n] = 0.f;
  if (n == 0) *locmax = KEY_NEG_INF;
}

// ---------------------------------------------------------------------------
// Edge kernel (hot path): each wave32 owns TPW 16-edge tiles; B (M_w rows
// 0..63, f16) is register-resident across tiles. Per tile:
//   msg(16x32) = [z[src] | z[dst]] (16x64 f16) @ B   -> 4x v_wmma_f32_16x16x32_f16
//   epilogue:  += M_b + ef * M_w[64]   (f32), then branchless atomic
//   segment-max into encoded u_keys.
// ---------------------------------------------------------------------------
__global__ void k_edge(const _Float16* __restrict__ zh, const float* __restrict__ ef,
                       const int* __restrict__ src, const int* __restrict__ dst,
                       const _Float16* __restrict__ Bh, const float* __restrict__ M_b,
                       const float* __restrict__ M_w,
                       unsigned* __restrict__ ukeys, long E) {
  const int wave = (blockIdx.x * blockDim.x + threadIdx.x) >> 5;
  const int lane = threadIdx.x & 31;
  const int hh   = lane >> 4;   // lane half selects K sub-block
  const int m    = lane & 15;   // row within tile / output column
  const long nTiles = (E + 15) >> 4;
  const long tile0 = (long)wave * TPW;
  if (tile0 >= nTiles) return;  // wave-uniform

  // --- B matrix: 2 K-chunks x 2 N-tiles, register resident ---
  v16h b[2][2];
#pragma unroll
  for (int c = 0; c < 2; ++c)
#pragma unroll
    for (int tl = 0; tl < 2; ++tl) {
      const _Float16* p = Bh + (((c * 2 + tl) * 32) + lane) * 16;
      v8h lo = *(const v8h*)p;
      v8h hi = *(const v8h*)(p + 8);
      b[c][tl] = __builtin_shufflevector(lo, hi, 0,1,2,3,4,5,6,7,8,9,10,11,12,13,14,15);
    }
  const float mb0 = M_b[m];
  const float mb1 = M_b[16 + m];
  const float w64_0 = M_w[64 * HID + m];        // edge-feature weight, cols 0..15
  const float w64_1 = M_w[64 * HID + 16 + m];   // cols 16..31

  for (int tt = 0; tt < TPW; ++tt) {
    const long tile = tile0 + tt;
    if (tile >= nTiles) break;  // wave-uniform
    const long tileBase = tile * 16;
    long e = tileBase + m; if (e >= E) e = E - 1;  // clamp tail (max idempotent)
    const int s = src[e];
    const int d = dst[e];

    // A chunk 0: z[src] row (K 0..31); lane<16 -> K{0..15}, lane>=16 -> K{16..31}
    const _Float16* rs = zh + (size_t)s * HID;
    v8h a0lo = *(const v8h*)(rs + hh * 16);
    v8h a0hi = *(const v8h*)(rs + hh * 16 + 8);
    v16h a0 = __builtin_shufflevector(a0lo, a0hi, 0,1,2,3,4,5,6,7,8,9,10,11,12,13,14,15);
    // A chunk 1: z[dst] row (K 32..63)
    const _Float16* rd = zh + (size_t)d * HID;
    v8h a1lo = *(const v8h*)(rd + hh * 16);
    v8h a1hi = *(const v8h*)(rd + hh * 16 + 8);
    v16h a1 = __builtin_shufflevector(a1lo, a1hi, 0,1,2,3,4,5,6,7,8,9,10,11,12,13,14,15);

    v8f c0 = {};  // cols 0..15
    v8f c1 = {};  // cols 16..31  (chains interleaved: 1 independent WMMA between deps)
    c0 = __builtin_amdgcn_wmma_f32_16x16x32_f16(false, a0, false, b[0][0], (short)0, c0, false, false);
    c1 = __builtin_amdgcn_wmma_f32_16x16x32_f16(false, a0, false, b[0][1], (short)0, c1, false, false);
    c0 = __builtin_amdgcn_wmma_f32_16x16x32_f16(false, a1, false, b[1][0], (short)0, c0, false, false);
    c1 = __builtin_amdgcn_wmma_f32_16x16x32_f16(false, a1, false, b[1][1], (short)0, c1, false, false);

    // Epilogue: D element r -> edge row (tileBase + r + 8*hh), col m / m+16.
    const long rbase = tileBase + hh * 8;
    if (tileBase + 16 <= E) {  // full tile: vector loads of dst ids + edge feats
      int4  di0 = *(const int4*)(dst + rbase);
      int4  di1 = *(const int4*)(dst + rbase + 4);
      float4 ef0 = *(const float4*)(ef + rbase);
      float4 ef1 = *(const float4*)(ef + rbase + 4);
      int   nd[8] = { di0.x, di0.y, di0.z, di0.w, di1.x, di1.y, di1.z, di1.w };
      float ev[8] = { ef0.x, ef0.y, ef0.z, ef0.w, ef1.x, ef1.y, ef1.z, ef1.w };
#pragma unroll
      for (int r = 0; r < 8; ++r) {
        float v0 = c0[r] + mb0 + ev[r] * w64_0;
        float v1 = c1[r] + mb1 + ev[r] * w64_1;
        atomicMax(&ukeys[(size_t)nd[r] * HID + m],      f2key(v0));
        atomicMax(&ukeys[(size_t)nd[r] * HID + 16 + m], f2key(v1));
      }
    } else {  // tail tile (at most one wave): clamped scalar path
#pragma unroll
      for (int r = 0; r < 8; ++r) {
        long er = rbase + r; if (er >= E) er = E - 1;
        int node = dst[er];
        float evr = ef[er];
        float v0 = c0[r] + mb0 + evr * w64_0;
        float v1 = c1[r] + mb1 + evr * w64_1;
        atomicMax(&ukeys[(size_t)node * HID + m],      f2key(v0));
        atomicMax(&ukeys[(size_t)node * HID + 16 + m], f2key(v1));
      }
    }
  }
}

// ---------------------------------------------------------------------------
// Per-step update: new_hidden, nne, new_state, loc-max, h-sum (f32 VALU)
// ---------------------------------------------------------------------------
__global__ void k_update(const float* __restrict__ z, const unsigned* __restrict__ ukeys,
                         const float* __restrict__ U_w, const float* __restrict__ U_b,
                         const float* __restrict__ dn_w, const float* __restrict__ dn_b,
                         const float* __restrict__ du_w, const float* __restrict__ du_b,
                         const float* __restrict__ term_w, const float* __restrict__ term_b,
                         float* __restrict__ hidden, float* __restrict__ preds_next,
                         float* __restrict__ nextnode, float* __restrict__ hsum,
                         unsigned int* __restrict__ locmax, int N, int t, int TM1) {
  __shared__ float Ul[64 * HID];
  __shared__ float dnl[64], dul[65], tml[HID], Ubl[HID], hs[HID];
  for (int i = threadIdx.x; i < 64 * HID; i += blockDim.x) Ul[i] = U_w[i];
  if (threadIdx.x < 64) dnl[threadIdx.x] = dn_w[threadIdx.x];
  if (threadIdx.x < 65) dul[threadIdx.x] = du_w[threadIdx.x];
  if (threadIdx.x < HID) {
    tml[threadIdx.x] = term_w[threadIdx.x];
    Ubl[threadIdx.x] = U_b[threadIdx.x];
    hs[threadIdx.x] = 0.f;
  }
  __syncthreads();
  int n = blockIdx.x * blockDim.x + threadIdx.x;
  if (n < N) {
    float zv[HID], nh[HID];
#pragma unroll
    for (int k = 0; k < HID; ++k) zv[k] = z[n * HID + k];
#pragma unroll
    for (int c = 0; c < HID; ++c) nh[c] = Ubl[c];
    for (int k = 0; k < HID; ++k) {
      float v = zv[k];
#pragma unroll
      for (int c = 0; c < HID; ++c) nh[c] = fmaf(v, Ul[k * HID + c], nh[c]);
    }
    for (int k = 0; k < HID; ++k) {
      float v = key2f(ukeys[n * HID + k]);   // decode segment-max
#pragma unroll
      for (int c = 0; c < HID; ++c) nh[c] = fmaf(v, Ul[(HID + k) * HID + c], nh[c]);
    }
#pragma unroll
    for (int c = 0; c < HID; ++c) hidden[n * HID + c] = nh[c];

    float nne = dn_b[0];
#pragma unroll
    for (int k = 0; k < HID; ++k) nne = fmaf(nh[k], dnl[k], fmaf(zv[k], dnl[HID + k], nne));
    nextnode[(size_t)n * TM1 + t] = nne;

    float ns = du_b[0];
#pragma unroll
    for (int k = 0; k < HID; ++k) ns = fmaf(nh[k], dul[k], fmaf(zv[k], dul[HID + k], ns));
    ns = fmaf(nne, dul[2 * HID], ns);
    preds_next[n] = ns;

    float loc = term_b[0];
#pragma unroll
    for (int k = 0; k < HID; ++k) loc = fmaf(nh[k], tml[k], loc);
    atomicMax(locmax, f2key(loc));
#pragma unroll
    for (int c = 0; c < HID; ++c) atomicAdd(&hs[c], nh[c]);
  }
  __syncthreads();
  if (threadIdx.x < HID) atomicAdd(&hsum[threadIdx.x], hs[threadIdx.x]);
}

// ---------------------------------------------------------------------------
// Per-step finalize: stop = sigmoid(max(max_n loc, loc(h_mean)))
// ---------------------------------------------------------------------------
__global__ void k_finalize(const float* __restrict__ hsum, const unsigned int* __restrict__ locmax,
                           const float* __restrict__ term_w, const float* __restrict__ term_b,
                           float* __restrict__ stop_out, int N) {
  int l = threadIdx.x;
  float p = (hsum[l] / (float)N) * term_w[l];
  for (int off = 16; off > 0; off >>= 1) p += __shfl_down(p, off, 32);
  if (l == 0) {
    float mx = fmaxf(key2f(*locmax), p + term_b[0]);
    stop_out[0] = 1.f / (1.f + expf(-mx));
  }
}

// ---------------------------------------------------------------------------
extern "C" void kernel_launch(void* const* d_in, const int* in_sizes, int n_in,
                              void* d_out, int out_size, void* d_ws, size_t ws_size,
                              hipStream_t stream) {
  const float* states    = (const float*)d_in[0];
  const float* priority  = (const float*)d_in[1];
  const float* edges_mat = (const float*)d_in[2];
  const int*   src       = (const int*)d_in[3];
  const int*   dst       = (const int*)d_in[4];
  const float* enc_w  = (const float*)d_in[5];
  const float* enc_b  = (const float*)d_in[6];
  const float* M_w    = (const float*)d_in[7];
  const float* M_b    = (const float*)d_in[8];
  const float* U_w    = (const float*)d_in[9];
  const float* U_b    = (const float*)d_in[10];
  const float* dn_w   = (const float*)d_in[11];
  const float* dn_b   = (const float*)d_in[12];
  const float* du_w   = (const float*)d_in[13];
  const float* du_b   = (const float*)d_in[14];
  const float* term_w = (const float*)d_in[15];
  const float* term_b = (const float*)d_in[16];

  const int  N = in_sizes[1];
  const int  T = in_sizes[0] / N;
  const long E = in_sizes[3];

  float* preds    = (float*)d_out;              // (T, N)
  float* stops    = preds + (size_t)T * N;      // (T,)
  float* nextnode = stops + T;                  // (N, T-1)

  // workspace carve-out (256B aligned)
  char* ws = (char*)d_ws;
  size_t off = 0;
  auto carve = [&](size_t bytes) { size_t o = off; off = (off + bytes + 255) & ~(size_t)255; return o; };
  float*     z      = (float*)(ws + carve((size_t)N * HID * 4));
  _Float16*  zh     = (_Float16*)(ws + carve((size_t)N * HID * 2));
  float*     hidden = (float*)(ws + carve((size_t)N * HID * 4));
  unsigned*  ukeys  = (unsigned*)(ws + carve((size_t)N * HID * 4));
  float*     ef     = (float*)(ws + carve((size_t)E * 4));
  _Float16*  Bh     = (_Float16*)(ws + carve(2 * 2 * 32 * 16 * 2));
  float*     hsum   = (float*)(ws + carve(HID * 4));
  unsigned*  locmax = (unsigned*)(ws + carve(4));
  (void)ws_size; (void)out_size; (void)n_in;

  const int TB = 256;
  k_setup<<<(N * HID + TB - 1) / TB, TB, 0, stream>>>(states, hidden, preds, stops, N);
  k_edgefeat<<<(unsigned)((E + TB - 1) / TB), TB, 0, stream>>>(edges_mat, src, dst, ef, E, N);
  k_packB<<<1, 128, 0, stream>>>(M_w, Bh);

  const long nTiles = (E + 15) / 16;
  const long nWaves = (nTiles + TPW - 1) / TPW;
  const unsigned edgeBlocks = (unsigned)((nWaves + 3) / 4); // 128 threads = 4 waves
  const unsigned nodeBlocks = (unsigned)((N + TB - 1) / TB);

  for (int t = 0; t < T - 1; ++t) {
    k_encode<<<nodeBlocks, TB, 0, stream>>>(preds + (size_t)t * N, hidden, priority,
                                            enc_w, enc_b, z, zh, ukeys, hsum, locmax, N);
    k_edge<<<edgeBlocks, 128, 0, stream>>>(zh, ef, src, dst, Bh, M_b, M_w, ukeys, E);
    k_update<<<nodeBlocks, TB, 0, stream>>>(z, ukeys, U_w, U_b, dn_w, dn_b, du_w, du_b,
                                            term_w, term_b, hidden, preds + (size_t)(t + 1) * N,
                                            nextnode, hsum, locmax, N, t, T - 1);
    k_finalize<<<1, 32, 0, stream>>>(hsum, locmax, term_w, term_b, stops + (t + 1), N);
  }
}
